// ScoreNet_73383811219609
// MI455X (gfx1250) — matrix operand order repaired and checked by reading, not compiled, not run on hardware
//
#include <hip/hip_runtime.h>
#include <hip/hip_bf16.h>

// ---------------------------------------------------------------------------
// GCN-style ScoreNet for MI455X (gfx1250, wave32).
// Dense linear layers via v_wmma_f32_16x16x32_f16 with pre-packed per-lane
// fragment weights (contiguous b128 loads, 2 M-tiles/wave); edge aggregation
// via edge-packed waves (32/nch edges per wave) with coalesced gathers and
// global float atomics. Whole working set is L2-resident (node features
// 6.4MB + edge data ~38MB << 192MB L2), so the scatter runs at L2 rates.
// All channel dims padded to 32 with zeros (self-maintaining under tanh/relu).
// ---------------------------------------------------------------------------

typedef __attribute__((ext_vector_type(16))) _Float16 v16h;
typedef __attribute__((ext_vector_type(8)))  _Float16 v8h;
typedef __attribute__((ext_vector_type(8)))  float    v8f;

#define NCH 32
static constexpr float kDelta = 2.0f;

// ---------------- utility kernels ----------------

__global__ void k_zero(float* __restrict__ p, long n) {
    long i = blockIdx.x * (long)blockDim.x + threadIdx.x;
    if (i < n) p[i] = 0.0f;
}

__global__ void k_pack_x(const float* __restrict__ x, _Float16* __restrict__ H, long n) {
    long i = blockIdx.x * (long)blockDim.x + threadIdx.x;
    if (i < n) H[i] = (_Float16)x[i];
}

// Pack fi x fo fp32 weight directly into WMMA B-fragment order:
// Wf[tile(2)][lane(32)][elem(16)] f16, where for lane L, elem i:
//   k = i<8 ? (L>>4)*8 + i : 16 + (L>>4)*8 + (i-8),  n = tile*16 + (L&15);  zero-padded to 32x32.
// GEMM then loads each lane's fragment as one contiguous 32B chunk.
__global__ void k_pack_w(const float* __restrict__ W, int fi, int fo,
                         _Float16* __restrict__ Wf) {
    int i = blockIdx.x * blockDim.x + threadIdx.x;
    if (i >= 2 * 32 * 16) return;
    int t    = i >> 9;          // n-tile
    int lane = (i >> 4) & 31;
    int e    = i & 15;
    int k = (e < 8) ? (lane >> 4) * 8 + e : 16 + (lane >> 4) * 8 + (e - 8);
    int n = t * 16 + (lane & 15);
    Wf[i] = (k < fi && n < fo) ? (_Float16)W[k * fo + n] : (_Float16)0.0f;
}

__global__ void k_pack_b(const float* __restrict__ b, int fo, float* __restrict__ bp) {
    int i = threadIdx.x;
    if (i < NCH) bp[i] = (i < fo) ? b[i] : 0.0f;
}

// ---------------- degree / edge normalization ----------------

__global__ void k_deg(const int* __restrict__ src, const float* __restrict__ w,
                      float* __restrict__ deg, int E) {
    int e = blockIdx.x * blockDim.x + threadIdx.x;
    if (e < E) atomicAdd(&deg[src[e]], w[e]);
}

__global__ void k_dis(float* __restrict__ deg, int n) {
    int i = blockIdx.x * blockDim.x + threadIdx.x;
    if (i < n) {
        float d = deg[i];
        deg[i] = (d > 0.0f) ? rsqrtf(fmaxf(d, 1e-30f)) : 0.0f;
    }
}

__global__ void k_normw(const int* __restrict__ src, const int* __restrict__ dst,
                        const float* __restrict__ w, const float* __restrict__ dis,
                        float* __restrict__ nw, int E) {
    int e = blockIdx.x * blockDim.x + threadIdx.x;
    if (e < E) nw[e] = dis[src[e]] * w[e] * dis[dst[e]];
}

// ---------------- WMMA GEMM helpers ----------------

// Load A fragment for a 16-row tile: two aligned 16B chunks per lane.
__device__ __forceinline__ v16h load_a_frag(const _Float16* __restrict__ H,
                                            long row0, int mrow, int half) {
    const v8h* hp8 = (const v8h*)(H + (row0 + mrow) * NCH);
    v8h alo = hp8[half];        // K = half*8 .. +7
    v8h ahi = hp8[2 + half];    // K = 16 + half*8 .. +7
    return __builtin_shufflevector(alo, ahi, 0, 1, 2, 3, 4, 5, 6, 7,
                                   8, 9, 10, 11, 12, 13, 14, 15);
}

// T[N,32] = H[N,32] @ W(32,32); 2 row-tiles per wave, B fragments reused.
__global__ void k_gemm_wmma(const _Float16* __restrict__ H,
                            const _Float16* __restrict__ Wf,
                            float* __restrict__ T, int ntiles) {
    int wid  = blockIdx.x * (blockDim.x >> 5) + (threadIdx.x >> 5);
    int lane = threadIdx.x & 31;
    int half = lane >> 4;
    int mrow = lane & 15;
    int ncol = lane & 15;

    v16h b0 = *(const v16h*)(Wf + (size_t)lane * 16);
    v16h b1 = *(const v16h*)(Wf + (size_t)(32 + lane) * 16);

#pragma unroll
    for (int t = 0; t < 2; ++t) {
        int tile = wid * 2 + t;
        if (tile >= ntiles) return;
        long row0 = (long)tile * 16;
        v16h a = load_a_frag(H, row0, mrow, half);
        v8f c0 = {}, c1 = {};
        c0 = __builtin_amdgcn_wmma_f32_16x16x32_f16(false, a, false, b0,
                                                    (short)0, c0, false, false);
        c1 = __builtin_amdgcn_wmma_f32_16x16x32_f16(false, a, false, b1,
                                                    (short)0, c1, false, false);
#pragma unroll
        for (int r = 0; r < 8; ++r) {
            int m = r + 8 * half;
            T[(row0 + m) * NCH + ncol]      = c0[r];
            T[(row0 + m) * NCH + 16 + ncol] = c1[r];
        }
    }
}

// MLP layer fused: H = relu(H @ W + b), f16 in/out, in place (each wave
// reads only the rows it writes; all loads precede stores).
__global__ void k_gemm_wmma_relu(_Float16* __restrict__ H,
                                 const _Float16* __restrict__ Wf,
                                 const float* __restrict__ bp, int ntiles) {
    int wid  = blockIdx.x * (blockDim.x >> 5) + (threadIdx.x >> 5);
    int lane = threadIdx.x & 31;
    int half = lane >> 4;
    int mrow = lane & 15;
    int ncol = lane & 15;

    v16h b0 = *(const v16h*)(Wf + (size_t)lane * 16);
    v16h b1 = *(const v16h*)(Wf + (size_t)(32 + lane) * 16);
    float bias0 = bp[ncol];
    float bias1 = bp[16 + ncol];

#pragma unroll
    for (int t = 0; t < 2; ++t) {
        int tile = wid * 2 + t;
        if (tile >= ntiles) return;
        long row0 = (long)tile * 16;
        v16h a = load_a_frag(H, row0, mrow, half);
        v8f c0 = {}, c1 = {};
        c0 = __builtin_amdgcn_wmma_f32_16x16x32_f16(false, a, false, b0,
                                                    (short)0, c0, false, false);
        c1 = __builtin_amdgcn_wmma_f32_16x16x32_f16(false, a, false, b1,
                                                    (short)0, c1, false, false);
#pragma unroll
        for (int r = 0; r < 8; ++r) {
            int m = r + 8 * half;
            H[(row0 + m) * NCH + ncol]      = (_Float16)fmaxf(c0[r] + bias0, 0.0f);
            H[(row0 + m) * NCH + 16 + ncol] = (_Float16)fmaxf(c1[r] + bias1, 0.0f);
        }
    }
}

// ---------------- edge scatter: Agg[dst] += nw * T[src] ----------------
// Edge-packed waves: 32/nch edges per wave. Lane sub-group handles one edge,
// lane-within-group = channel. Gathers and atomics stay coalesced per group.
// lg = log2(nch) in {3,4,5}.

__global__ void k_scatter(const float* __restrict__ T, const int* __restrict__ src,
                          const int* __restrict__ dst, const float* __restrict__ nw,
                          float* __restrict__ Agg, int E, int lg) {
    int wave = blockIdx.x * (blockDim.x >> 5) + (threadIdx.x >> 5);
    int lane = threadIdx.x & 31;
    int nch  = 1 << lg;
    int sub  = lane >> lg;                 // edge slot within wave
    int c    = lane & (nch - 1);           // channel
    int e    = wave * (32 >> lg) + sub;
    if (e >= E) return;
    int   s = src[e];
    int   d = dst[e];
    float w = nw[e];
    const float* srow = T + (long)s * NCH;
    __builtin_prefetch(srow, 0, 1);        // global_prefetch_b8 (gfx1250)
    atomicAdd(&Agg[(long)d * NCH + c], w * srow[c]);
}

// ---------------- combine stage (MP): consume Agg and re-zero it ----------------

__global__ void k_combine_mp(const float* __restrict__ T, float* __restrict__ Agg,
                             const float* __restrict__ bp, _Float16* __restrict__ H,
                             long n) {
    long i = blockIdx.x * (long)blockDim.x + threadIdx.x;
    if (i >= n) return;
    int c = (int)(i & 31);
    float v = (1.0f - kDelta) * T[i] + kDelta * Agg[i] + bp[c];
    Agg[i] = 0.0f;                       // ready for next layer's scatter
    H[i] = (_Float16)tanhf(v);
}

__global__ void k_final(const _Float16* __restrict__ H, const float* __restrict__ fw,
                        const float* __restrict__ fb, float* __restrict__ out, int n) {
    int i = blockIdx.x * blockDim.x + threadIdx.x;
    if (i >= n) return;
    const _Float16* hp = H + (long)i * NCH;
    float acc = fb[0];
#pragma unroll
    for (int c = 0; c < 16; ++c) acc += (float)hp[c] * fw[c];
    out[i] = tanhf(acc);
}

// ---------------- host orchestration ----------------

static inline size_t align256(size_t x) { return (x + 255) & ~(size_t)255; }

static inline int ilog2(int v) { int l = 0; while ((1 << l) < v) ++l; return l; }

extern "C" void kernel_launch(void* const* d_in, const int* in_sizes, int n_in,
                              void* d_out, int out_size, void* d_ws, size_t ws_size,
                              hipStream_t stream) {
    const int N = in_sizes[0] / NCH;      // 50000
    const int E = in_sizes[2];            // 3,200,000 directed edges

    const float* x   = (const float*)d_in[0];
    const int*   ei  = (const int*)d_in[1];
    const int*   src = ei;                // edge_index row 0
    const int*   dst = ei + E;            // edge_index row 1
    const float* ew  = (const float*)d_in[2];
    const float* fin_w = (const float*)d_in[31];
    const float* fin_b = (const float*)d_in[32];
    (void)n_in; (void)ws_size;

    // Workspace carve-up
    char* p = (char*)d_ws;
    _Float16* H   = (_Float16*)p;  p += align256((size_t)N * NCH * sizeof(_Float16));
    float*    T   = (float*)p;     p += align256((size_t)N * NCH * sizeof(float));
    float*    Agg = (float*)p;     p += align256((size_t)N * NCH * sizeof(float));
    float*    dis = (float*)p;     p += align256((size_t)N * sizeof(float));
    float*    nw  = (float*)p;     p += align256((size_t)E * sizeof(float));
    _Float16* Wf  = (_Float16*)p;  p += align256((size_t)14 * NCH * NCH * sizeof(_Float16));
    float*    bp  = (float*)p;     p += align256((size_t)14 * NCH * sizeof(float));

    const long NE32   = (long)N * NCH;
    const int  ntiles = N / 16;                       // 3125 full 16-row tiles
    const int  nwaves = (ntiles + 1) / 2;             // 2 tiles per wave
    const int  gElem  = (int)((NE32 + 255) / 256);
    const int  gEdge  = (E + 255) / 256;
    const int  gGemm  = (nwaves + 7) / 8;

    // Layer dims: [32] + MP_UNITS, then MLP 8->16->16
    const int dims[13] = {32, 32, 32, 32, 32, 16, 16, 16, 16, 8, 8, 8, 8};

    // Pack activations + all weights/biases into padded fragment blocks.
    k_pack_x<<<gElem, 256, 0, stream>>>(x, H, NE32);
    for (int l = 0; l < 12; ++l) {
        k_pack_w<<<4, 256, 0, stream>>>((const float*)d_in[3 + l], dims[l], dims[l + 1],
                                        Wf + (size_t)l * NCH * NCH);
        k_pack_b<<<1, 32, 0, stream>>>((const float*)d_in[15 + l], dims[l + 1],
                                       bp + (size_t)l * NCH);
    }
    const int mlp_fi[2] = {8, 16};
    const int mlp_fo[2] = {16, 16};
    for (int m = 0; m < 2; ++m) {
        k_pack_w<<<4, 256, 0, stream>>>((const float*)d_in[27 + m], mlp_fi[m], mlp_fo[m],
                                        Wf + (size_t)(12 + m) * NCH * NCH);
        k_pack_b<<<1, 32, 0, stream>>>((const float*)d_in[29 + m], mlp_fo[m],
                                       bp + (size_t)(12 + m) * NCH);
    }

    // Symmetric normalization: nw = dis[src] * w * dis[dst]
    k_zero<<<(N + 255) / 256, 256, 0, stream>>>(dis, N);
    k_deg<<<gEdge, 256, 0, stream>>>(src, ew, dis, E);
    k_dis<<<(N + 255) / 256, 256, 0, stream>>>(dis, N);
    k_normw<<<gEdge, 256, 0, stream>>>(src, dst, ew, dis, nw, E);

    // Agg zeroed once; k_combine_mp re-zeros it each layer after consuming.
    k_zero<<<gElem, 256, 0, stream>>>(Agg, NE32);

    // 12 message-passing layers
    for (int l = 0; l < 12; ++l) {
        const int lg  = ilog2(dims[l + 1]);           // log2(out channels)
        const int epw = 32 >> lg;                     // edges per wave
        const int waves = (E + epw - 1) / epw;
        const int gScat = (waves + 7) / 8;            // 8 waves / 256-thread block
        k_gemm_wmma<<<gGemm, 256, 0, stream>>>(H, Wf + (size_t)l * NCH * NCH, T, ntiles);
        k_scatter<<<gScat, 256, 0, stream>>>(T, src, dst, nw, Agg, E, lg);
        k_combine_mp<<<gElem, 256, 0, stream>>>(T, Agg, bp + (size_t)l * NCH, H, NE32);
    }

    // 2 MLP layers fused (GEMM + bias + ReLU, f16 in-place)
    for (int m = 0; m < 2; ++m) {
        k_gemm_wmma_relu<<<gGemm, 256, 0, stream>>>(H, Wf + (size_t)(12 + m) * NCH * NCH,
                                                    bp + (size_t)(12 + m) * NCH, ntiles);
    }

    // Final scalar head + tanh
    k_final<<<(N + 255) / 256, 256, 0, stream>>>(H, fin_w, fin_b, (float*)d_out, N);
    (void)out_size;
}
